// ArgExtractorLayer_35527969472569
// MI455X (gfx1250) — matrix operand order repaired and checked by reading, not compiled, not run on hardware
//
#include <hip/hip_runtime.h>
#include <hip/hip_bf16.h>
#include <stdint.h>

// ---------------- problem constants ----------------
#define L_   1024
#define B_   4
#define H_   8
#define DH_  96
#define D_   768          // H_*DH_
#define DFF_ 2048
#define U_   35           // FACTOR * ceil(ln(1024)) = 5*7
#define NT_  (L_*B_)      // 4096 tokens
#define LN_EPS 1e-5f

typedef __bf16 bf16_t;
typedef bf16_t v16bf __attribute__((ext_vector_type(16)));
typedef float  v8f   __attribute__((ext_vector_type(8)));

union FragBf { v16bf v; unsigned int u[8]; };

// ---------------- CDNA5 async global->LDS DMA helpers ----------------
// GLOBAL_LOAD_ASYNC_TO_LDS_B128 (§15.18.3 op 98): per-lane 16B DMA into LDS,
// tracked with ASYNCcnt (completes in order for loads).
__device__ __forceinline__ uint32_t lds_lo32(const void* p) {
    // generic pointer to LDS: addr[31:0] is the LDS byte offset
    return (uint32_t)(uintptr_t)p;
}
__device__ __forceinline__ void async_ld_b128(uint32_t lds, const void* gaddr) {
    asm volatile("global_load_async_to_lds_b128 %0, %1, off"
                 :: "v"(lds), "v"(gaddr) : "memory");
}
__device__ __forceinline__ void wait_async_le8() {
    asm volatile("s_wait_asynccnt 0x8" ::: "memory");
}
__device__ __forceinline__ void wait_async_le0() {
    asm volatile("s_wait_asynccnt 0x0" ::: "memory");
}

// ---------------- small helpers ----------------
__device__ __forceinline__ unsigned int samp_hash(int l, int j) {
    unsigned int x = (unsigned int)l * 2654435761u + (unsigned int)j * 0x9E3779B9u + 12345u;
    x ^= x >> 16; x *= 0x85EBCA6Bu; x ^= x >> 13;
    return x & (L_ - 1);
}

// ---------------- K0: fp32 -> bf16 convert ----------------
__global__ void cvt_bf16_kernel(const float* __restrict__ src, bf16_t* __restrict__ dst, int n) {
    int i = blockIdx.x * blockDim.x + threadIdx.x;
    if (i < n) dst[i] = (bf16_t)src[i];
}

// ---------------- K1: sampled QK scores, M = max - mean ----------------
__global__ void attn_sample_scores(const float* __restrict__ target,
                                   const float* __restrict__ source,
                                   float* __restrict__ M) {
    int t = blockIdx.x * blockDim.x + threadIdx.x;
    if (t >= B_ * H_ * L_) return;
    int l = t % L_;
    int h = (t / L_) % H_;
    int b = t / (L_ * H_);
    const float* q = target + ((size_t)l * B_ + b) * D_ + h * DH_;
    float mx = -__builtin_inff();
    float sum = 0.f;
    for (int j = 0; j < U_; ++j) {
        int kl = (int)samp_hash(l, j);
        const float* k = source + ((size_t)kl * B_ + b) * D_ + h * DH_;
        float s = 0.f;
        for (int d = 0; d < DH_; ++d) s += q[d] * k[d];
        mx = fmaxf(mx, s);
        sum += s;
    }
    M[((size_t)b * H_ + h) * L_ + l] = mx - sum * (1.f / U_);
}

// ---------------- K2: per-(b,h) top-U selection + scatter map ----------------
__global__ __launch_bounds__(256) void topk_kernel(const float* __restrict__ M,
                                                   int* __restrict__ top,
                                                   int* __restrict__ ovr) {
    int bh = blockIdx.x;                 // 0 .. B_*H_-1
    __shared__ float vals[L_];
    __shared__ float rv[256];
    __shared__ int   ri[256];
    for (int i = threadIdx.x; i < L_; i += 256) {
        vals[i] = M[(size_t)bh * L_ + i];
        ovr[(size_t)bh * L_ + i] = -1;
    }
    __syncthreads();
    for (int it = 0; it < U_; ++it) {
        float best = -__builtin_inff(); int bi = 0;
        for (int i = threadIdx.x; i < L_; i += 256) {
            float v = vals[i];
            if (v > best) { best = v; bi = i; }
        }
        rv[threadIdx.x] = best; ri[threadIdx.x] = bi;
        __syncthreads();
        for (int s = 128; s > 0; s >>= 1) {
            if (threadIdx.x < s && rv[threadIdx.x + s] > rv[threadIdx.x]) {
                rv[threadIdx.x] = rv[threadIdx.x + s];
                ri[threadIdx.x] = ri[threadIdx.x + s];
            }
            __syncthreads();
        }
        if (threadIdx.x == 0) {
            int idx = ri[0];
            top[bh * U_ + it] = idx;
            vals[idx] = -__builtin_inff();
            ovr[(size_t)bh * L_ + idx] = it;
        }
        __syncthreads();
    }
}

// ---------------- K2b: V mean over L ----------------
__global__ void vmean_kernel(const float* __restrict__ source, float* __restrict__ vmean) {
    int t = blockIdx.x * blockDim.x + threadIdx.x;
    if (t >= B_ * H_ * DH_) return;
    int d = t % DH_;
    int h = (t / DH_) % H_;
    int b = t / (DH_ * H_);
    float s = 0.f;
    for (int l = 0; l < L_; ++l) s += source[((size_t)l * B_ + b) * D_ + h * DH_ + d];
    vmean[t] = s * (1.f / L_);
}

// ---------------- K3: attention for the U reduced queries ----------------
__global__ __launch_bounds__(128) void reduced_attention(const float* __restrict__ target,
                                                         const float* __restrict__ source,
                                                         const int* __restrict__ top,
                                                         float* __restrict__ upd) {
    int gid = blockIdx.x;                 // B_*H_*U_
    int ui = gid % U_;
    int h  = (gid / U_) % H_;
    int b  = gid / (U_ * H_);
    int bh = b * H_ + h;
    __shared__ float sc[L_];
    __shared__ float red[128];
    int lq = top[bh * U_ + ui];
    const float scale = rsqrtf((float)DH_);
    const float* q = target + ((size_t)lq * B_ + b) * D_ + h * DH_;
    for (int l = threadIdx.x; l < L_; l += 128) {
        const float* k = source + ((size_t)l * B_ + b) * D_ + h * DH_;
        float s = 0.f;
        for (int d = 0; d < DH_; ++d) s += q[d] * k[d];
        sc[l] = s * scale;
    }
    __syncthreads();
    float mx = -__builtin_inff();
    for (int l = threadIdx.x; l < L_; l += 128) mx = fmaxf(mx, sc[l]);
    red[threadIdx.x] = mx; __syncthreads();
    for (int s = 64; s > 0; s >>= 1) {
        if (threadIdx.x < s) red[threadIdx.x] = fmaxf(red[threadIdx.x], red[threadIdx.x + s]);
        __syncthreads();
    }
    mx = red[0];
    __syncthreads();
    float sum = 0.f;
    for (int l = threadIdx.x; l < L_; l += 128) {
        float e = __expf(sc[l] - mx);
        sc[l] = e;
        sum += e;
    }
    red[threadIdx.x] = sum; __syncthreads();
    for (int s = 64; s > 0; s >>= 1) {
        if (threadIdx.x < s) red[threadIdx.x] += red[threadIdx.x + s];
        __syncthreads();
    }
    float inv = 1.f / red[0];
    for (int d = threadIdx.x; d < DH_; d += 128) {
        float a = 0.f;
        for (int l = 0; l < L_; ++l)
            a += sc[l] * source[((size_t)l * B_ + b) * D_ + h * DH_ + d];
        upd[((size_t)bh * U_ + ui) * DH_ + d] = a * inv;
    }
}

// ---------------- K4: scatter/broadcast + residual + LN1 ----------------
__global__ __launch_bounds__(256) void add_attn_ln1(const float* __restrict__ target,
                                                    const float* __restrict__ upd,
                                                    const float* __restrict__ vmean,
                                                    const int* __restrict__ ovr,
                                                    const float* __restrict__ g,
                                                    const float* __restrict__ beta,
                                                    float* __restrict__ out_f,
                                                    bf16_t* __restrict__ out_b) {
    int r = blockIdx.x;                    // token = l*B_+b
    int b = r % B_;
    int l = r / B_;
    __shared__ float red[256];
    float x[3];
#pragma unroll
    for (int i = 0; i < 3; ++i) {
        int d  = threadIdx.x + i * 256;
        int h  = d / DH_, dd = d % DH_;
        int bh = b * H_ + h;
        int ov = ovr[(size_t)bh * L_ + l];
        float av = (ov >= 0) ? upd[((size_t)bh * U_ + ov) * DH_ + dd]
                             : vmean[bh * DH_ + dd];
        x[i] = target[(size_t)r * D_ + d] + av;
    }
    float s = x[0] + x[1] + x[2];
    red[threadIdx.x] = s; __syncthreads();
    for (int st = 128; st > 0; st >>= 1) {
        if (threadIdx.x < st) red[threadIdx.x] += red[threadIdx.x + st];
        __syncthreads();
    }
    float mu = red[0] * (1.f / D_);
    __syncthreads();
    float vs = 0.f;
#pragma unroll
    for (int i = 0; i < 3; ++i) { float dl = x[i] - mu; vs += dl * dl; }
    red[threadIdx.x] = vs; __syncthreads();
    for (int st = 128; st > 0; st >>= 1) {
        if (threadIdx.x < st) red[threadIdx.x] += red[threadIdx.x + st];
        __syncthreads();
    }
    float rstd = rsqrtf(red[0] * (1.f / D_) + LN_EPS);
#pragma unroll
    for (int i = 0; i < 3; ++i) {
        int d = threadIdx.x + i * 256;
        float y = (x[i] - mu) * rstd * g[d] + beta[d];
        out_f[(size_t)r * D_ + d] = y;
        out_b[(size_t)r * D_ + d] = (bf16_t)y;
    }
}

// ---------------- K5/K6: WMMA bf16 GEMM  C = A(MxK) * W(NxK)^T ----------------
// Double-buffered LDS staged with async global->LDS DMA (ASYNCcnt pipelining).
// EPI==0: relu(x + bias) -> bf16 out ; EPI==1: x + bias -> f32 out
#define BM 128
#define BN 128
#define BK 64

template <int EPI>
__global__ __launch_bounds__(256) void gemm_wmma_bf16(const bf16_t* __restrict__ A,
                                                      const bf16_t* __restrict__ W,
                                                      const float* __restrict__ bias,
                                                      void* __restrict__ Cout,
                                                      int Mdim, int Ndim, int Kdim) {
    __shared__ __align__(16) bf16_t As[2][BM * BK];   // [m][k], k contiguous
    __shared__ __align__(16) bf16_t Bs[2][BN * BK];   // [n][k], k contiguous
    const int tid  = threadIdx.x;
    const int lane = tid & 31;
    const int wave = tid >> 5;           // 0..7
    const int wm   = wave & 1;           // 2 waves along M
    const int wn   = wave >> 1;          // 4 waves along N
    const int m0   = blockIdx.x * BM;
    const int n0   = blockIdx.y * BN;
    const int wrow = wm * 64;            // wave tile: 64 (M) x 32 (N)
    const int wcol = wn * 32;
    const int half = lane >> 4;          // lane half selects K sub-ranges
    const int mn15 = lane & 15;

    // staging pattern: tile row = srow + 32*t (t=0..3), 8 bf16 (16B) at scol
    const int srow = tid >> 3;           // 0..31
    const int scol = (tid & 7) * 8;      // 0..56

    const bf16_t* gA = A + (size_t)(m0 + srow) * Kdim + scol;
    const bf16_t* gB = W + (size_t)(n0 + srow) * Kdim + scol;
    const uint32_t ldsA[2] = { lds_lo32(&As[0][srow * BK + scol]),
                               lds_lo32(&As[1][srow * BK + scol]) };
    const uint32_t ldsB[2] = { lds_lo32(&Bs[0][srow * BK + scol]),
                               lds_lo32(&Bs[1][srow * BK + scol]) };

    v8f acc[4][2];
#pragma unroll
    for (int i = 0; i < 4; ++i)
#pragma unroll
        for (int j = 0; j < 2; ++j) acc[i][j] = (v8f){};

    const int nk = Kdim / BK;

    // issue tile 0 (8 async b128 DMAs per thread: 4 A rows + 4 B rows)
#pragma unroll
    for (int t = 0; t < 4; ++t) {
        async_ld_b128(ldsA[0] + (uint32_t)(t * 32 * BK * 2), gA + (size_t)(32 * t) * Kdim);
        async_ld_b128(ldsB[0] + (uint32_t)(t * 32 * BK * 2), gB + (size_t)(32 * t) * Kdim);
    }

    for (int k = 0; k < nk; ++k) {
        const int cur = k & 1;
        if (k + 1 < nk) {
            const int nxt = cur ^ 1;
            const int kk  = (k + 1) * BK;
#pragma unroll
            for (int t = 0; t < 4; ++t) {
                async_ld_b128(ldsA[nxt] + (uint32_t)(t * 32 * BK * 2),
                              gA + (size_t)(32 * t) * Kdim + kk);
                async_ld_b128(ldsB[nxt] + (uint32_t)(t * 32 * BK * 2),
                              gB + (size_t)(32 * t) * Kdim + kk);
            }
            wait_async_le8();   // in-order completion: tile k's 8 DMAs are done
        } else {
            wait_async_le0();
        }
        __syncthreads();        // all threads' tile-k data resident in LDS

        const bf16_t* Ab = &As[cur][0];
        const bf16_t* Bb = &Bs[cur][0];
#pragma unroll
        for (int kt = 0; kt < BK; kt += 32) {
            // B fragments: lane -> col n = mn15; element p -> K pair half*16 + 2p
            FragBf bfr[2];
#pragma unroll
            for (int j = 0; j < 2; ++j) {
                const bf16_t* bp = &Bb[(wcol + j * 16 + mn15) * BK + kt + half * 16];
#pragma unroll
                for (int p = 0; p < 8; ++p)
                    bfr[j].u[p] = *(const unsigned int*)(bp + 2 * p);
            }
            // A fragments: lane -> row m = mn15; element p -> K pair half*16 + 2p
#pragma unroll
            for (int i = 0; i < 4; ++i) {
                FragBf afr;
                const bf16_t* ap = &Ab[(wrow + i * 16 + mn15) * BK + kt];
#pragma unroll
                for (int p = 0; p < 4; ++p) {
                    afr.u[p]     = *(const unsigned int*)(ap + half * 16 + 2 * p);
                    afr.u[4 + p] = *(const unsigned int*)(ap + half * 16 + 8 + 2 * p);
                }
#pragma unroll
                for (int j = 0; j < 2; ++j)
                    acc[i][j] = __builtin_amdgcn_wmma_f32_16x16x32_bf16(
                        false, afr.v, false, bfr[j].v, (short)0, acc[i][j], false, false);
            }
        }
        __syncthreads();        // done reading LDS before tile k+2 overwrites it
    }

    // epilogue: D layout row = r + 8*half, col = mn15
#pragma unroll
    for (int i = 0; i < 4; ++i)
#pragma unroll
        for (int j = 0; j < 2; ++j) {
            int col = n0 + wcol + j * 16 + mn15;
            float bv = bias[col];
#pragma unroll
            for (int r = 0; r < 8; ++r) {
                int row = m0 + wrow + i * 16 + r + 8 * half;
                float v = acc[i][j][r] + bv;
                if (EPI == 0) {
                    v = v > 0.f ? v : 0.f;
                    ((bf16_t*)Cout)[(size_t)row * Ndim + col] = (bf16_t)v;
                } else {
                    ((float*)Cout)[(size_t)row * Ndim + col] = v;
                }
            }
        }
}

// ---------------- K7: residual + LN2 -> output ----------------
__global__ __launch_bounds__(256) void add_ln2(const float* __restrict__ n1,
                                               const float* __restrict__ y2,
                                               const float* __restrict__ g,
                                               const float* __restrict__ beta,
                                               float* __restrict__ out) {
    int r = blockIdx.x;
    __shared__ float red[256];
    float x[3];
#pragma unroll
    for (int i = 0; i < 3; ++i) {
        int d = threadIdx.x + i * 256;
        x[i] = n1[(size_t)r * D_ + d] + y2[(size_t)r * D_ + d];
    }
    float s = x[0] + x[1] + x[2];
    red[threadIdx.x] = s; __syncthreads();
    for (int st = 128; st > 0; st >>= 1) {
        if (threadIdx.x < st) red[threadIdx.x] += red[threadIdx.x + st];
        __syncthreads();
    }
    float mu = red[0] * (1.f / D_);
    __syncthreads();
    float vs = 0.f;
#pragma unroll
    for (int i = 0; i < 3; ++i) { float dl = x[i] - mu; vs += dl * dl; }
    red[threadIdx.x] = vs; __syncthreads();
    for (int st = 128; st > 0; st >>= 1) {
        if (threadIdx.x < st) red[threadIdx.x] += red[threadIdx.x + st];
        __syncthreads();
    }
    float rstd = rsqrtf(red[0] * (1.f / D_) + LN_EPS);
#pragma unroll
    for (int i = 0; i < 3; ++i) {
        int d = threadIdx.x + i * 256;
        out[(size_t)r * D_ + d] = (x[i] - mu) * rstd * g[d] + beta[d];
    }
}

// ---------------- host: kernel_launch ----------------
extern "C" void kernel_launch(void* const* d_in, const int* in_sizes, int n_in,
                              void* d_out, int out_size, void* d_ws, size_t ws_size,
                              hipStream_t stream) {
    const float* target = (const float*)d_in[0];   // (L,B,D)
    const float* source = (const float*)d_in[1];   // (L,B,D)
    const float* W1     = (const float*)d_in[2];   // (DFF,D)
    const float* b1     = (const float*)d_in[3];   // (DFF)
    const float* W2     = (const float*)d_in[4];   // (D,DFF)
    const float* b2     = (const float*)d_in[5];   // (D)
    const float* ln1g   = (const float*)d_in[6];
    const float* ln1b   = (const float*)d_in[7];
    const float* ln2g   = (const float*)d_in[8];
    const float* ln2b   = (const float*)d_in[9];
    float* out = (float*)d_out;

    // workspace carve-up (256B aligned)
    char* p = (char*)d_ws;
    auto alloc = [&](size_t bytes) -> void* {
        void* r = (void*)p;
        p += (bytes + 255) & ~(size_t)255;
        return r;
    };
    float*  Mbuf   = (float*)alloc((size_t)B_ * H_ * L_ * sizeof(float));
    int*    top    = (int*)  alloc((size_t)B_ * H_ * U_ * sizeof(int));
    int*    ovr    = (int*)  alloc((size_t)B_ * H_ * L_ * sizeof(int));
    float*  vmean  = (float*)alloc((size_t)B_ * H_ * DH_ * sizeof(float));
    float*  upd    = (float*)alloc((size_t)B_ * H_ * U_ * DH_ * sizeof(float));
    float*  n1f    = (float*)alloc((size_t)NT_ * D_ * sizeof(float));
    bf16_t* n1b    = (bf16_t*)alloc((size_t)NT_ * D_ * sizeof(bf16_t));
    bf16_t* W1b    = (bf16_t*)alloc((size_t)DFF_ * D_ * sizeof(bf16_t));
    bf16_t* W2b    = (bf16_t*)alloc((size_t)D_ * DFF_ * sizeof(bf16_t));
    bf16_t* H1     = (bf16_t*)alloc((size_t)NT_ * DFF_ * sizeof(bf16_t));
    float*  y2     = (float*)alloc((size_t)NT_ * D_ * sizeof(float));
    (void)ws_size; (void)n_in; (void)in_sizes; (void)out_size;

    // weight conversion (every call; deterministic)
    {
        int n = DFF_ * D_;
        cvt_bf16_kernel<<<(n + 255) / 256, 256, 0, stream>>>(W1, W1b, n);
        cvt_bf16_kernel<<<(n + 255) / 256, 256, 0, stream>>>(W2, W2b, n);
    }
    // attention
    attn_sample_scores<<<(B_ * H_ * L_ + 255) / 256, 256, 0, stream>>>(target, source, Mbuf);
    topk_kernel<<<B_ * H_, 256, 0, stream>>>(Mbuf, top, ovr);
    vmean_kernel<<<(B_ * H_ * DH_ + 255) / 256, 256, 0, stream>>>(source, vmean);
    reduced_attention<<<B_ * H_ * U_, 128, 0, stream>>>(target, source, top, upd);
    add_attn_ln1<<<NT_, 256, 0, stream>>>(target, upd, vmean, ovr, ln1g, ln1b, n1f, n1b);
    // FFN (WMMA, async-staged LDS double buffering)
    gemm_wmma_bf16<0><<<dim3(NT_ / BM, DFF_ / BN), 256, 0, stream>>>(n1b, W1b, b1, (void*)H1, NT_, DFF_, D_);
    gemm_wmma_bf16<1><<<dim3(NT_ / BM, D_ / BN), 256, 0, stream>>>(H1, W2b, b2, (void*)y2, NT_, D_, DFF_);
    // LN2 -> out
    add_ln2<<<NT_, 256, 0, stream>>>(n1f, y2, ln2g, ln2b, out);
}
